// VerticalModel_GT_GIN_CombinedEmbeds_51118700757168
// MI455X (gfx1250) — matrix operand order repaired and checked by reading, not compiled, not run on hardware
//
#include <hip/hip_runtime.h>
#include <stdint.h>

// ---------------------------------------------------------------------------
// Problem constants (match reference)
// ---------------------------------------------------------------------------
#define NN  20000
#define EE  320000
#define GG  512
#define FF  128
#define HH  4
#define CC  64
#define HC  256
#define EDD 16

typedef __attribute__((ext_vector_type(16))) _Float16 v16h;
typedef __attribute__((ext_vector_type(8)))  float    v8f;
typedef __attribute__((ext_vector_type(4)))  unsigned uint4v;

union HFrag { uint4v u[2]; v16h v; };

// ---------------------------------------------------------------------------
// CDNA5 async global->LDS helpers (ASYNCcnt path, ISA ch.10 async ops)
// ---------------------------------------------------------------------------
__device__ __forceinline__ void async_g2l_b128(void* lds_ptr, const void* gptr) {
  unsigned lds_off = (unsigned)(uintptr_t)lds_ptr;       // low 32 bits = LDS offset
  unsigned long long ga = (unsigned long long)(uintptr_t)gptr;
  asm volatile("global_load_async_to_lds_b128 %0, %1, off"
               :: "v"(lds_off), "v"(ga) : "memory");
}
__device__ __forceinline__ void wait_async0() {
  asm volatile("s_wait_asynccnt 0" ::: "memory");
}

// ---------------------------------------------------------------------------
// WMMA GEMM:  Y[M,Nout] = X[M,K] (f16) * Wt[Nout,K]^T (f16, pre-transposed)
//             + bias, optional ReLU.  fp32 accumulate.
// One wave owns a 16x64 strip: 4 accumulators, 4 v_wmma per K-step.
// Requires: M%16==0, K%32==0, Nout%64==0 (all call sites satisfy this).
// ---------------------------------------------------------------------------
__global__ __launch_bounds__(256) void wmma_gemm_kernel(
    const _Float16* __restrict__ X, const _Float16* __restrict__ Wt,
    const float* __restrict__ bias, float* __restrict__ Y,
    int M, int K, int Nout, int relu)
{
  const int lane = threadIdx.x & 31;
  const int wave = threadIdx.x >> 5;
  const int half = lane >> 4;      // 0: lanes 0-15, 1: lanes 16-31
  const int l16  = lane & 15;
  const int nGroups = Nout >> 6;
  const int mTiles  = M >> 4;
  int wid = blockIdx.x * 8 + wave;
  if (wid >= mTiles * nGroups) return;
  int mt = wid / nGroups;
  int ng = wid - mt * nGroups;
  int m0 = mt << 4, n0 = ng << 6;

  v8f acc0 = {}, acc1 = {}, acc2 = {}, acc3 = {};
  const _Float16* xrow = X + (size_t)(m0 + l16) * K;
  const _Float16* b0r  = Wt + (size_t)(n0      + l16) * K;
  const _Float16* b1r  = Wt + (size_t)(n0 + 16 + l16) * K;
  const _Float16* b2r  = Wt + (size_t)(n0 + 32 + l16) * K;
  const _Float16* b3r  = Wt + (size_t)(n0 + 48 + l16) * K;

  for (int k0 = 0; k0 < K; k0 += 32) {
    HFrag fa, f0, f1, f2, f3;
    // A fragment (16x32 f16): lanes<16 carry K {0..15}, lanes>=16 {16..31}
    const _Float16* ap = xrow + k0 + half * 16;
    __builtin_prefetch(ap + 32, 0, 1);           // next K tile -> global_prefetch
    fa.u[0] = *(const uint4v*)(ap);
    fa.u[1] = *(const uint4v*)(ap + 8);
    // B fragments (32x16 f16): lane n = l16, K = k0 + half*16 .. +15 contiguous in Wt row
    const int bo = k0 + half * 16;
    f0.u[0] = *(const uint4v*)(b0r + bo); f0.u[1] = *(const uint4v*)(b0r + bo + 8);
    f1.u[0] = *(const uint4v*)(b1r + bo); f1.u[1] = *(const uint4v*)(b1r + bo + 8);
    f2.u[0] = *(const uint4v*)(b2r + bo); f2.u[1] = *(const uint4v*)(b2r + bo + 8);
    f3.u[0] = *(const uint4v*)(b3r + bo); f3.u[1] = *(const uint4v*)(b3r + bo + 8);
    acc0 = __builtin_amdgcn_wmma_f32_16x16x32_f16(false, fa.v, false, f0.v, (short)0, acc0, false, false);
    acc1 = __builtin_amdgcn_wmma_f32_16x16x32_f16(false, fa.v, false, f1.v, (short)0, acc1, false, false);
    acc2 = __builtin_amdgcn_wmma_f32_16x16x32_f16(false, fa.v, false, f2.v, (short)0, acc2, false, false);
    acc3 = __builtin_amdgcn_wmma_f32_16x16x32_f16(false, fa.v, false, f3.v, (short)0, acc3, false, false);
  }
  // C layout: VGPR r -> row m0 + r + 8*half, col n0 + l16
  #pragma unroll
  for (int r = 0; r < 8; ++r) {
    int row = m0 + r + (half << 3);
    float* yr = Y + (size_t)row * Nout + n0 + l16;
    float vals[4] = {acc0[r], acc1[r], acc2[r], acc3[r]};
    #pragma unroll
    for (int t = 0; t < 4; ++t) {
      float o = vals[t] + (bias ? bias[n0 + t * 16 + l16] : 0.f);
      if (relu) o = fmaxf(o, 0.f);
      yr[t * 16] = o;
    }
  }
}

// ---------------------------------------------------------------------------
// Small utility kernels
// ---------------------------------------------------------------------------
__global__ void fill_f32_kernel(float* p, float v, int n) {
  int t = blockIdx.x * 256 + threadIdx.x; if (t < n) p[t] = v;
}
__global__ void fill_u32_kernel(unsigned* p, unsigned v, int n) {
  int t = blockIdx.x * 256 + threadIdx.x; if (t < n) p[t] = v;
}
__global__ void f32_to_f16_kernel(const float* __restrict__ s, _Float16* __restrict__ d, int n) {
  int t = blockIdx.x * 256 + threadIdx.x; if (t < n) d[t] = (_Float16)s[t];
}
// W[din,dout] f32  ->  Wt[dout,din] f16
__global__ void transpose_f16_kernel(const float* __restrict__ W, _Float16* __restrict__ Wt,
                                     int din, int dout) {
  int t = blockIdx.x * 256 + threadIdx.x;
  if (t < din * dout) {
    int k = t / dout, n = t - k * dout;
    Wt[(size_t)n * din + k] = (_Float16)W[t];
  }
}
__global__ void add_kernel(const float* __restrict__ a, const float* __restrict__ b,
                           float* __restrict__ c, int n) {
  int t = blockIdx.x * 256 + threadIdx.x; if (t < n) c[t] = a[t] + b[t];
}

// ---------------------------------------------------------------------------
// GT edge kernels (one thread per (edge, head)); We staged to LDS via the
// CDNA5 async global->LDS path (one 16KB copy per block instead of per-thread
// rereads of the whole table).
// ---------------------------------------------------------------------------
#define WE_DWORDS (EDD * HC)   // 4096 floats = 16KB

__device__ __forceinline__ unsigned enc_f32(float f) {
  unsigned u = __float_as_uint(f);
  return (u & 0x80000000u) ? ~u : (u | 0x80000000u);
}
__device__ __forceinline__ float dec_f32(unsigned e) {
  unsigned u = (e & 0x80000000u) ? (e ^ 0x80000000u) : ~e;
  return __uint_as_float(u);
}

__global__ __launch_bounds__(256) void edge_score_kernel(
    const float* __restrict__ q, const float* __restrict__ k,
    const float* __restrict__ ea, const float* __restrict__ We,
    const int* __restrict__ src, const int* __restrict__ dst,
    float* __restrict__ sc, unsigned* __restrict__ mx)
{
  __shared__ float sWe[WE_DWORDS];
  for (int i = threadIdx.x; i < WE_DWORDS / 4; i += 256)
    async_g2l_b128(&sWe[i * 4], We + i * 4);
  wait_async0();
  __syncthreads();

  int t = blockIdx.x * 256 + threadIdx.x;
  if (t < EE * HH) {
    int e = t >> 2, h = t & 3;
    int s = src[e], d = dst[e];
    float a[EDD];
    #pragma unroll
    for (int i = 0; i < EDD; ++i) a[i] = ea[e * EDD + i];
    const float* qp = q + (size_t)d * HC + h * CC;
    const float* kp = k + (size_t)s * HC + h * CC;
    const float* wp = sWe + h * CC;
    float sum = 0.f;
    for (int c = 0; c < CC; ++c) {
      float eh = 0.f;
      #pragma unroll
      for (int i = 0; i < EDD; ++i) eh += a[i] * wp[i * HC + c];
      sum += qp[c] * (kp[c] + eh);
    }
    float score = sum * 0.125f;              // 1/sqrt(64)
    sc[t] = score;
    atomicMax(mx + (size_t)d * HH + h, enc_f32(score));
  }
}

__global__ void edge_exp_kernel(float* __restrict__ sc, const unsigned* __restrict__ mx,
                                const int* __restrict__ dst, float* __restrict__ dn)
{
  int t = blockIdx.x * 256 + threadIdx.x;
  if (t >= EE * HH) return;
  int e = t >> 2, h = t & 3;
  int d = dst[e];
  float ex = __expf(sc[t] - dec_f32(mx[(size_t)d * HH + h]));
  sc[t] = ex;
  atomicAdd(dn + (size_t)d * HH + h, ex);
}

__global__ __launch_bounds__(256) void edge_agg_kernel(
    const float* __restrict__ sc, const float* __restrict__ dn,
    const float* __restrict__ v, const float* __restrict__ ea,
    const float* __restrict__ We, const int* __restrict__ src,
    const int* __restrict__ dst, float* __restrict__ out)
{
  __shared__ float sWe[WE_DWORDS];
  for (int i = threadIdx.x; i < WE_DWORDS / 4; i += 256)
    async_g2l_b128(&sWe[i * 4], We + i * 4);
  wait_async0();
  __syncthreads();

  int t = blockIdx.x * 256 + threadIdx.x;
  if (t < EE * HH) {
    int e = t >> 2, h = t & 3;
    int s = src[e], d = dst[e];
    float alpha = sc[t] / (dn[(size_t)d * HH + h] + 1e-16f);
    float a[EDD];
    #pragma unroll
    for (int i = 0; i < EDD; ++i) a[i] = ea[e * EDD + i];
    const float* vp = v + (size_t)s * HC + h * CC;
    const float* wp = sWe + h * CC;
    float* op = out + (size_t)d * HC + h * CC;
    for (int c = 0; c < CC; ++c) {
      float eh = 0.f;
      #pragma unroll
      for (int i = 0; i < EDD; ++i) eh += a[i] * wp[i * HC + c];
      atomicAdd(op + c, alpha * (vp[c] + eh));
    }
  }
}

// beta-gated skip combine (one thread per node); beta weights staged in LDS
__global__ __launch_bounds__(128) void gt_combine_kernel(
    const float* __restrict__ out, const float* __restrict__ sk,
    const float* __restrict__ bw, float* __restrict__ hn)
{
  __shared__ float sbw[3 * HC];
  for (int i = threadIdx.x; i < 3 * HC; i += 128) sbw[i] = bw[i];
  __syncthreads();

  int n = blockIdx.x * 128 + threadIdx.x;
  if (n >= NN) return;
  const float* o = out + (size_t)n * HC;
  const float* s = sk  + (size_t)n * HC;
  float acc = 0.f;
  for (int j = 0; j < HC; ++j) {
    float oj = o[j], sj = s[j];
    acc += oj * sbw[j] + sj * sbw[HC + j] + (oj - sj) * sbw[2 * HC + j];
  }
  float beta = 1.f / (1.f + __expf(-acc));
  float* hp = hn + (size_t)n * HC;
  for (int j = 0; j < HC; ++j) hp[j] = beta * s[j] + (1.f - beta) * o[j];
}

// ---------------------------------------------------------------------------
// Pooling / GIN / readout kernels
// ---------------------------------------------------------------------------
__global__ void pool_mean_kernel(const float* __restrict__ h, const int* __restrict__ batch,
                                 float* __restrict__ gtf, float* __restrict__ cnts)
{
  int t = blockIdx.x * 256 + threadIdx.x;
  if (t >= NN * HC) return;
  int n = t >> 8, j = t & 255;
  int b = batch[n];
  atomicAdd(gtf + (size_t)b * HC + j, h[t]);
  if (j == 0) atomicAdd(cnts + b, 1.f);
}
__global__ void pool_div_kernel(float* __restrict__ gtf, const float* __restrict__ cnts)
{
  int t = blockIdx.x * 256 + threadIdx.x;
  if (t >= GG * HC) return;
  gtf[t] /= fmaxf(cnts[t >> 8], 1.f);
}
__global__ void gin_agg_kernel(const float* __restrict__ g, const int* __restrict__ src,
                               const int* __restrict__ dst, float* __restrict__ agg, int Wd)
{
  long long t = (long long)blockIdx.x * 256 + threadIdx.x;
  if (t >= (long long)EE * Wd) return;
  int e = (int)(t / Wd);
  int c = (int)(t - (long long)e * Wd);
  atomicAdd(agg + (size_t)dst[e] * Wd + c, g[(size_t)src[e] * Wd + c]);
}
__global__ void pool_sum_kernel(const float* __restrict__ g, const int* __restrict__ batch,
                                float* __restrict__ ginf)
{
  int t = blockIdx.x * 256 + threadIdx.x;
  if (t >= NN * CC) return;
  int n = t >> 6, c = t & 63;
  atomicAdd(ginf + (size_t)batch[n] * CC + c, g[t]);
}
__global__ void concat_kernel(const float* __restrict__ gtf, const float* __restrict__ ginf,
                              float* __restrict__ z)
{
  int t = blockIdx.x * 256 + threadIdx.x;
  if (t >= GG * (HC + CC)) return;
  int g = t / (HC + CC), j = t - g * (HC + CC);
  z[t] = (j < HC) ? gtf[(size_t)g * HC + j] : ginf[(size_t)g * CC + (j - HC)];
}
__global__ void ro2_kernel(const float* __restrict__ zr, const float* __restrict__ w,
                           const float* __restrict__ b, float* __restrict__ out)
{
  int g = blockIdx.x * 128 + threadIdx.x;
  if (g >= GG) return;
  float s = b[0];
  for (int c = 0; c < CC; ++c) s += zr[(size_t)g * CC + c] * w[c];
  out[g] = s;
}

// ---------------------------------------------------------------------------
// Host-side orchestration
// ---------------------------------------------------------------------------
static inline int cdiv_ll(long long a, int b) { return (int)((a + b - 1) / b); }

static void run_gemm(hipStream_t st, const float* W, const float* bias,
                     const _Float16* Xh, _Float16* Wt, float* Y,
                     int M, int K, int Nout, int relu)
{
  transpose_f16_kernel<<<cdiv_ll((long long)K * Nout, 256), 256, 0, st>>>(W, Wt, K, Nout);
  int waves = (M / 16) * (Nout / 64);
  wmma_gemm_kernel<<<(waves + 7) / 8, 256, 0, st>>>(Xh, Wt, bias, Y, M, K, Nout, relu);
}

extern "C" void kernel_launch(void* const* d_in, const int* in_sizes, int n_in,
                              void* d_out, int out_size, void* d_ws, size_t ws_size,
                              hipStream_t stream)
{
  (void)in_sizes; (void)n_in; (void)out_size; (void)ws_size;
  // Input order: insertion-order flatten of setup_inputs()
  const float* x  = (const float*)d_in[0];
  const float* ea = (const float*)d_in[1];
  // gt layer l base = 2 + 10*l : qw qb kw kb vw vb ew sw sb betaw
  // gin layer l base = 32 + 4*l : l1w l1b l2w l2b
  const float* ro1w = (const float*)d_in[44];
  const float* ro1b = (const float*)d_in[45];
  const float* ro2w = (const float*)d_in[46];
  const float* ro2b = (const float*)d_in[47];
  const int* ei    = (const int*)d_in[48];
  const int* batch = (const int*)d_in[49];
  const int* src = ei;
  const int* dst = ei + EE;
  float* outp = (float*)d_out;

  // ---- workspace carve-out ----
  char* wsp = (char*)d_ws;
  auto carve = [&](size_t bytes) -> void* {
    void* p = wsp; wsp += (bytes + 255) & ~(size_t)255; return p;
  };
  const size_t NB = (size_t)NN * HC;           // 5.12M elems
  float*    h_a  = (float*)carve(NB * 4);
  float*    h_b  = (float*)carve(NB * 4);
  float*    qB   = (float*)carve(NB * 4);
  float*    kB   = (float*)carve(NB * 4);
  float*    vB   = (float*)carve(NB * 4);
  float*    sB   = (float*)carve(NB * 4);
  float*    ob   = (float*)carve(NB * 4);      // attn out / GIN agg
  float*    tb   = (float*)carve(NB * 4);      // GIN t
  _Float16* hb16 = (_Float16*)carve(NB * 2);   // f16 GEMM input staging
  _Float16* wt16 = (_Float16*)carve((size_t)HC * 320 * 2);  // transposed weights
  float*    sc   = (float*)carve((size_t)EE * HH * 4);
  unsigned* mxu  = (unsigned*)carve((size_t)NN * HH * 4);
  float*    dn   = (float*)carve((size_t)NN * HH * 4);
  float*    g_a  = (float*)carve((size_t)NN * CC * 4);
  float*    g_b  = (float*)carve((size_t)NN * CC * 4);
  float*    t2   = (float*)carve((size_t)NN * CC * 4);
  float*    gtf  = (float*)carve((size_t)GG * HC * 4);
  float*    ginf = (float*)carve((size_t)GG * CC * 4);
  float*    cnts = (float*)carve((size_t)GG * 4);
  float*    z    = (float*)carve((size_t)GG * (HC + CC) * 4);
  float*    zr   = (float*)carve((size_t)GG * CC * 4);

  const int EH = EE * HH;

  // ================= GT layers =================
  const float* hin = x;
  int din = FF;
  float* hbufs[2] = {h_a, h_b};
  for (int l = 0; l < 3; ++l) {
    const float* qw = (const float*)d_in[2 + 10 * l + 0];
    const float* qb = (const float*)d_in[2 + 10 * l + 1];
    const float* kw = (const float*)d_in[2 + 10 * l + 2];
    const float* kb = (const float*)d_in[2 + 10 * l + 3];
    const float* vw = (const float*)d_in[2 + 10 * l + 4];
    const float* vb = (const float*)d_in[2 + 10 * l + 5];
    const float* ew = (const float*)d_in[2 + 10 * l + 6];
    const float* sw = (const float*)d_in[2 + 10 * l + 7];
    const float* sb = (const float*)d_in[2 + 10 * l + 8];
    const float* bw = (const float*)d_in[2 + 10 * l + 9];

    f32_to_f16_kernel<<<cdiv_ll((long long)NN * din, 256), 256, 0, stream>>>(hin, hb16, NN * din);
    run_gemm(stream, qw, qb, hb16, wt16, qB, NN, din, HC, 0);
    run_gemm(stream, kw, kb, hb16, wt16, kB, NN, din, HC, 0);
    run_gemm(stream, vw, vb, hb16, wt16, vB, NN, din, HC, 0);
    run_gemm(stream, sw, sb, hb16, wt16, sB, NN, din, HC, 0);

    fill_u32_kernel<<<cdiv_ll(NN * HH, 256), 256, 0, stream>>>(mxu, 0u, NN * HH);
    fill_f32_kernel<<<cdiv_ll(NN * HH, 256), 256, 0, stream>>>(dn, 0.f, NN * HH);
    edge_score_kernel<<<cdiv_ll(EH, 256), 256, 0, stream>>>(qB, kB, ea, ew, src, dst, sc, mxu);
    edge_exp_kernel<<<cdiv_ll(EH, 256), 256, 0, stream>>>(sc, mxu, dst, dn);
    fill_f32_kernel<<<cdiv_ll((long long)NB, 256), 256, 0, stream>>>(ob, 0.f, (int)NB);
    edge_agg_kernel<<<cdiv_ll(EH, 256), 256, 0, stream>>>(sc, dn, vB, ea, ew, src, dst, ob);

    float* hnext = hbufs[l & 1];
    gt_combine_kernel<<<cdiv_ll(NN, 128), 128, 0, stream>>>(ob, sB, bw, hnext);
    hin = hnext;
    din = HC;
  }
  const float* hfin = hin;   // h_a after 3 layers

  // ================= graph mean pool of GT output =================
  fill_f32_kernel<<<cdiv_ll(GG * HC, 256), 256, 0, stream>>>(gtf, 0.f, GG * HC);
  fill_f32_kernel<<<cdiv_ll(GG, 256), 256, 0, stream>>>(cnts, 0.f, GG);
  pool_mean_kernel<<<cdiv_ll((long long)NN * HC, 256), 256, 0, stream>>>(hfin, batch, gtf, cnts);
  pool_div_kernel<<<cdiv_ll(GG * HC, 256), 256, 0, stream>>>(gtf, cnts);

  // ================= GIN layers =================
  const float* g = hfin;
  int Wd = HC;
  float* gbufs[2] = {g_a, g_b};
  for (int l = 0; l < 3; ++l) {
    const float* l1w = (const float*)d_in[32 + 4 * l + 0];
    const float* l1b = (const float*)d_in[32 + 4 * l + 1];
    const float* l2w = (const float*)d_in[32 + 4 * l + 2];
    const float* l2b = (const float*)d_in[32 + 4 * l + 3];

    fill_f32_kernel<<<cdiv_ll((long long)NN * Wd, 256), 256, 0, stream>>>(ob, 0.f, NN * Wd);
    gin_agg_kernel<<<cdiv_ll((long long)EE * Wd, 256), 256, 0, stream>>>(g, src, dst, ob, Wd);
    add_kernel<<<cdiv_ll((long long)NN * Wd, 256), 256, 0, stream>>>(g, ob, tb, NN * Wd);
    f32_to_f16_kernel<<<cdiv_ll((long long)NN * Wd, 256), 256, 0, stream>>>(tb, hb16, NN * Wd);
    run_gemm(stream, l1w, l1b, hb16, wt16, t2, NN, Wd, CC, 1);
    f32_to_f16_kernel<<<cdiv_ll((long long)NN * CC, 256), 256, 0, stream>>>(t2, hb16, NN * CC);
    float* gn = gbufs[l & 1];
    run_gemm(stream, l2w, l2b, hb16, wt16, gn, NN, CC, CC, 0);
    g = gn;
    Wd = CC;
  }

  // ================= graph sum pool of GIN output =================
  fill_f32_kernel<<<cdiv_ll(GG * CC, 256), 256, 0, stream>>>(ginf, 0.f, GG * CC);
  pool_sum_kernel<<<cdiv_ll((long long)NN * CC, 256), 256, 0, stream>>>(g, batch, ginf);

  // ================= readout =================
  concat_kernel<<<cdiv_ll(GG * (HC + CC), 256), 256, 0, stream>>>(gtf, ginf, z);
  f32_to_f16_kernel<<<cdiv_ll(GG * (HC + CC), 256), 256, 0, stream>>>(z, hb16, GG * (HC + CC));
  run_gemm(stream, ro1w, ro1b, hb16, wt16, zr, GG, HC + CC, CC, 1);
  ro2_kernel<<<cdiv_ll(GG, 128), 128, 0, stream>>>(zr, ro2w, ro2b, outp);
}
